// LinearAttention_31181462569475
// MI455X (gfx1250) — matrix operand order repaired and checked by reading, not compiled, not run on hardware
//
#include <hip/hip_runtime.h>

typedef __attribute__((ext_vector_type(16))) _Float16 v16h;
typedef __attribute__((ext_vector_type(8)))  _Float16 h8;
typedef __attribute__((ext_vector_type(8)))  float    v8f;
typedef __attribute__((ext_vector_type(4)))  float    f4;   // native vec4 for NT ld/st

#define NTOK 8192
#define DIM  128
#define RMS_EPS 1.1920929e-07f

__device__ __forceinline__ float wave_reduce_sum(float v) {
#pragma unroll
  for (int m = 16; m >= 1; m >>= 1) v += __shfl_xor(v, m, 32);
  return v;
}

__device__ __forceinline__ float silu_f(float x) {
  return x * (1.0f / (1.0f + __expf(-x)));
}

// One block = 16 tokens, 256 threads (8 waves, wave32).
__global__ __launch_bounds__(256)
void linatt_fused_kernel(const float* __restrict__ x,
                         const float* __restrict__ past_mem,
                         const float* __restrict__ gamma,
                         const float* __restrict__ w_qkv,
                         const float* __restrict__ w_forget,
                         float* __restrict__ retrieved,
                         float* __restrict__ mem_out) {
  __shared__ __align__(16) _Float16 xnh[16 * DIM];    // fp16 RMSNorm'd acts (WMMA A)
  __shared__ __align__(16) float    qkv[16 * 384];    // q|k|v per token; q,k silu'd in place
  __shared__ float fgate[16];

  const int tid  = threadIdx.x;
  const int lane = tid & 31;
  const int w    = tid >> 5;          // wave id 0..7
  const int blk  = blockIdx.x;        // 0..511

  // ---------------- Phase 1: RMSNorm + forget gate (2 rows per wave) -------
  {
    const float4 g4  = *(const float4*)(gamma    + 4 * lane);
    const float4 wf4 = *(const float4*)(w_forget + 4 * lane);
#pragma unroll
    for (int rr = 0; rr < 2; ++rr) {
      const int r   = w * 2 + rr;
      const int tok = blk * 16 + r;
      const float4 xv = *(const float4*)(x + (size_t)tok * DIM + 4 * lane);
      float ss = xv.x * xv.x + xv.y * xv.y + xv.z * xv.z + xv.w * xv.w;
      ss = wave_reduce_sum(ss);
      const float rms = rsqrtf(ss * (1.0f / 128.0f) + RMS_EPS);
      float4 xn;
      xn.x = xv.x * rms * g4.x;
      xn.y = xv.y * rms * g4.y;
      xn.z = xv.z * rms * g4.z;
      xn.w = xv.w * rms * g4.w;
      xnh[r * DIM + 4 * lane + 0] = (_Float16)xn.x;
      xnh[r * DIM + 4 * lane + 1] = (_Float16)xn.y;
      xnh[r * DIM + 4 * lane + 2] = (_Float16)xn.z;
      xnh[r * DIM + 4 * lane + 3] = (_Float16)xn.w;
      float fd = xn.x * wf4.x + xn.y * wf4.y + xn.z * wf4.z + xn.w * wf4.w;
      fd = wave_reduce_sum(fd);
      if (lane == 0) fgate[r] = 1.0f / (1.0f + __expf(-fd));
    }
  }
  __syncthreads();

  // ---------------- Phase 2: QKV = xn @ w_qkv^T via WMMA f16 ---------------
  // C tiles: 16 rows (tokens) x 16 cols; 24 col tiles, 3 per wave; K = 4*32.
  {
    const int n  = lane & 15;   // tile column / A row (M)
    const int hi = lane >> 4;   // lane-half selector
#pragma unroll
    for (int t = 0; t < 3; ++t) {
      const int cb = (w + 8 * t) * 16;   // output column base in [0,384)
      v8f acc = {};
#pragma unroll
      for (int kc = 0; kc < 4; ++kc) {
        // A fragment (16x32 f16): lane holds row M=n; halves 0-7 -> K=kb..kb+7,
        // halves 8-15 -> K=kb+16..kb+23, kb = kc*32 + hi*8  (ISA 7.12.2)
        const _Float16* ap = &xnh[n * DIM + kc * 32 + hi * 8];
        union { v16h v; h8 h[2]; } au;
        au.h[0] = *(const h8*)(ap);
        au.h[1] = *(const h8*)(ap + 16);
        // B fragment (32x16 f16): lane holds col N=n, same K order as A (kb..kb+7, kb+16..kb+23).
        // B[k][n] = w_qkv[cb+n][k] -> contiguous run of w_qkv row (L2-resident).
        const float* wrow = w_qkv + (size_t)(cb + n) * DIM + kc * 32 + hi * 8;
        v16h b;
#pragma unroll
        for (int i = 0; i < 16; ++i) b[i] = (_Float16)wrow[i < 8 ? i : i + 8];
        acc = __builtin_amdgcn_wmma_f32_16x16x32_f16(
            /*neg_a=*/false, au.v, /*neg_b=*/false, b,
            /*c_mod=*/(short)0, acc, /*reuse_a=*/false, /*reuse_b=*/false);
      }
      // C/D layout: VGPR v -> row M = v + 8*hi, col N = n
#pragma unroll
      for (int v = 0; v < 8; ++v) qkv[(hi * 8 + v) * 384 + cb + n] = acc[v];
    }
  }
  __syncthreads();

  // ---------------- Phase 3: SiLU on q,k in place --------------------------
#pragma unroll
  for (int rr = 0; rr < 2; ++rr) {
    const int r = w * 2 + rr;
    const int e = 4 * lane;
#pragma unroll
    for (int j = 0; j < 4; ++j) {
      qkv[r * 384 + e + j]       = silu_f(qkv[r * 384 + e + j]);        // q
      qkv[r * 384 + 128 + e + j] = silu_f(qkv[r * 384 + 128 + e + j]);  // k
    }
  }
  __syncthreads();

  // ---------------- Phase 4: stream past_mem -> mem_out + retrieval --------
  // mem_new[d,e] = (1-f)*past[d,e] + (f*k[d])*v[e];  retr[e] += q[d]*mem_new
  // Lane owns columns 4*lane..4*lane+3 for its token: no cross-lane reduce.
  // past_mem / mem_out are touched exactly once (1 GiB stream >> 192 MB L2):
  // use non-temporal cache policy on both sides.
#pragma unroll
  for (int rr = 0; rr < 2; ++rr) {
    const int r   = w * 2 + rr;
    const int tok = blk * 16 + r;
    const float f = fgate[r];
    const float a = 1.0f - f;
    const float* qs = &qkv[r * 384];
    const float* ks = &qkv[r * 384 + 128];
    const float4 vv = *(const float4*)&qkv[r * 384 + 256 + 4 * lane];
    const float* pm = past_mem + (size_t)tok * DIM * DIM + 4 * lane;
    float*       mo = mem_out  + (size_t)tok * DIM * DIM + 4 * lane;
    float racc_x = 0.0f, racc_y = 0.0f, racc_z = 0.0f, racc_w = 0.0f;
#pragma unroll 2
    for (int d0 = 0; d0 < DIM; d0 += 4) {
      const float4 q4 = *(const float4*)(qs + d0);   // LDS broadcast, 1 op / 4 iters
      const float4 k4 = *(const float4*)(ks + d0);
      const float qd[4] = {q4.x, q4.y, q4.z, q4.w};
      const float kd[4] = {k4.x, k4.y, k4.z, k4.w};
#pragma unroll
      for (int j = 0; j < 4; ++j) {
        const int d = d0 + j;
        const float bk = f * kd[j];
        const f4 p = __builtin_nontemporal_load((const f4*)(pm + (size_t)d * DIM));
        f4 o;
        o.x = fmaf(bk, vv.x, a * p.x);
        o.y = fmaf(bk, vv.y, a * p.y);
        o.z = fmaf(bk, vv.z, a * p.z);
        o.w = fmaf(bk, vv.w, a * p.w);
        racc_x = fmaf(qd[j], o.x, racc_x);
        racc_y = fmaf(qd[j], o.y, racc_y);
        racc_z = fmaf(qd[j], o.z, racc_z);
        racc_w = fmaf(qd[j], o.w, racc_w);
        __builtin_nontemporal_store(o, (f4*)(mo + (size_t)d * DIM));
      }
    }
    float4 rout;
    rout.x = racc_x; rout.y = racc_y; rout.z = racc_z; rout.w = racc_w;
    *(float4*)(retrieved + (size_t)tok * DIM + 4 * lane) = rout;
  }
}

extern "C" void kernel_launch(void* const* d_in, const int* in_sizes, int n_in,
                              void* d_out, int out_size, void* d_ws, size_t ws_size,
                              hipStream_t stream) {
  (void)in_sizes; (void)n_in; (void)out_size; (void)d_ws; (void)ws_size;
  const float* x        = (const float*)d_in[0];
  const float* past_mem = (const float*)d_in[1];
  const float* gamma    = (const float*)d_in[2];
  const float* w_qkv    = (const float*)d_in[3];
  const float* w_forget = (const float*)d_in[4];
  float* retrieved = (float*)d_out;                      // [8192,128] first
  float* mem_out   = retrieved + (size_t)NTOK * DIM;     // then [8192,128,128]
  hipLaunchKernelGGL(linatt_fused_kernel, dim3(NTOK / 16), dim3(256), 0, stream,
                     x, past_mem, gamma, w_qkv, w_forget, retrieved, mem_out);
}
